// GNNLayer_89910845375343
// MI455X (gfx1250) — compile-verified, matrix operand order checked by reading, not run on hardware
//
#include <hip/hip_runtime.h>

#define N_NODES 50000
#define N_EDGES 800000
#define FEAT 96
#define EDGE_TILES (N_EDGES / 128)   // 6250 tiles of 128 edges
#define EDGE_GRID 625                // persistent blocks; ~10 tiles each

typedef __attribute__((ext_vector_type(16))) __bf16 v16bf;
typedef __attribute__((ext_vector_type(8)))  float  v8f;

union BF16Frag { unsigned int u[8]; v16bf v; };

// ---------------- workspace layout (bytes) ----------------
#define WS_DELTA 0ull          // 150000  f32
#define WS_AGG   600064ull     // 4800000 f32
#define WS_XBF   19800064ull   // 4800000 bf16
#define WS_W0F   29400064ull   // 128x128 bf16 (n-major, K padded, feature-permuted)
#define WS_W1F   29432832ull   // 96x128  bf16
#define WS_W0G   29457408ull   // 128x128 bf16
#define WS_W1G   29490176ull   // 96x128  bf16

__device__ __forceinline__ unsigned short bf16b(float f) {
  unsigned int u = __float_as_uint(f);
  return (unsigned short)((u + 0x7FFFu + ((u >> 16) & 1u)) >> 16);
}

// Load a 16x32 bf16 WMMA fragment for this lane from an LDS row (64 u32 = 128 bf16).
// v=0..3 and v=4..7 hit consecutive u32s -> compiler fuses into 2x ds_load_b128.
__device__ __forceinline__ v16bf ldfrag(const unsigned int* row, int kbase, int half) {
  BF16Frag f;
#pragma unroll
  for (int v = 0; v < 8; ++v) {
    int off = ((v & 4) ? 16 : 0) + half * 8 + 2 * (v & 3);
    f.u[v] = row[(kbase + off) >> 1];
  }
  return f.v;
}

// Deterministic float max via sign-split integer atomics.
__device__ __forceinline__ void atomicMaxF(float* addr, float val) {
  if (val >= 0.0f) atomicMax((int*)addr, __float_as_int(val));
  else             atomicMin((unsigned int*)addr, __float_as_uint(val));
}

// ---------------- prep: bf16 conversions, weight packing, agg init ----------------
__global__ void prep_kernel(const float* __restrict__ x,
                            const float* __restrict__ f_w0, const float* __restrict__ f_w1,
                            const float* __restrict__ g_w0, const float* __restrict__ g_w1,
                            char* __restrict__ ws) {
  long long i = (long long)blockIdx.x * blockDim.x + threadIdx.x;
  unsigned short* xbf  = (unsigned short*)(ws + WS_XBF);
  unsigned int*   aggu = (unsigned int*)(ws + WS_AGG);
  unsigned short* w0f  = (unsigned short*)(ws + WS_W0F);
  unsigned short* w1f  = (unsigned short*)(ws + WS_W1F);
  unsigned short* w0g  = (unsigned short*)(ws + WS_W0G);
  unsigned short* w1g  = (unsigned short*)(ws + WS_W1G);
  const long long NX = (long long)N_NODES * FEAT;
  if (i < NX) { xbf[i] = bf16b(x[i]); return; }
  i -= NX;
  if (i < NX) { aggu[i] = 0xFF800000u; return; }   // -inf
  i -= NX;
  if (i < 128 * 128) {  // mlp_f W0: e_in order [rel(3), x(96)] -> packed [x(96), rel(3), 0...]
    int n = (int)(i >> 7), k = (int)(i & 127);
    float v = (k < 96) ? f_w0[(k + 3) * 128 + n]
                       : ((k < 99) ? f_w0[(k - 96) * 128 + n] : 0.0f);
    w0f[n * 128 + k] = bf16b(v);
    return;
  }
  i -= 128 * 128;
  if (i < 96 * 128) {
    int n = (int)(i >> 7), k = (int)(i & 127);
    w1f[n * 128 + k] = bf16b(f_w1[k * 96 + n]);
    return;
  }
  i -= 96 * 128;
  if (i < 128 * 128) {
    int n = (int)(i >> 7), k = (int)(i & 127);
    w0g[n * 128 + k] = bf16b((k < 96) ? g_w0[k * 128 + n] : 0.0f);
    return;
  }
  i -= 128 * 128;
  if (i < 96 * 128) {
    int n = (int)(i >> 7), k = (int)(i & 127);
    w1g[n * 128 + k] = bf16b(g_w1[k * 96 + n]);
  }
}

// ---------------- delta = mlp_h(x): 96 -> 64 -> 3 (tiny, VALU) ----------------
__global__ void __launch_bounds__(256) delta_kernel(
    const float* __restrict__ x,
    const float* __restrict__ hw0, const float* __restrict__ hb0,
    const float* __restrict__ hw1, const float* __restrict__ hb1,
    char* __restrict__ ws) {
  __shared__ float sW0[96 * 64];
  __shared__ float sW1[64 * 3];
  __shared__ float sB0[64];
  __shared__ float sB1[3];
  for (int i = threadIdx.x; i < 96 * 64; i += 256) sW0[i] = hw0[i];
  for (int i = threadIdx.x; i < 64 * 3;  i += 256) sW1[i] = hw1[i];
  if (threadIdx.x < 64) sB0[threadIdx.x] = hb0[threadIdx.x];
  if (threadIdx.x < 3)  sB1[threadIdx.x] = hb1[threadIdx.x];
  __syncthreads();
  int n = blockIdx.x * 256 + threadIdx.x;
  if (n >= N_NODES) return;
  float hid[64];
#pragma unroll
  for (int j = 0; j < 64; ++j) hid[j] = sB0[j];
  const float* xr = x + (long long)n * FEAT;
  for (int k = 0; k < FEAT; ++k) {
    float xv = xr[k];
#pragma unroll
    for (int j = 0; j < 64; ++j) hid[j] = fmaf(xv, sW0[k * 64 + j], hid[j]);
  }
  float d0 = sB1[0], d1 = sB1[1], d2 = sB1[2];
#pragma unroll
  for (int j = 0; j < 64; ++j) {
    float h = fmaxf(hid[j], 0.0f);
    d0 = fmaf(h, sW1[j * 3 + 0], d0);
    d1 = fmaf(h, sW1[j * 3 + 1], d1);
    d2 = fmaf(h, sW1[j * 3 + 2], d2);
  }
  float* delta = (float*)(ws + WS_DELTA);
  delta[n * 3 + 0] = d0; delta[n * 3 + 1] = d1; delta[n * 3 + 2] = d2;
}

// ---------------- edge MLP (99->128->96) + scatter-max, WMMA, persistent ----------------
__global__ void __launch_bounds__(256) edge_kernel(
    const float* __restrict__ pos, const long long* __restrict__ ei,
    const float* __restrict__ fb0, const float* __restrict__ fb1,
    char* __restrict__ ws) {
  __shared__ unsigned int   sW0[128 * 64];       // W0f^T bf16 [128][128]
  __shared__ unsigned int   sW1[96 * 64];        // W1f^T bf16 [96][128]
  __shared__ float          sB0[128];
  __shared__ float          sB1[96];
  __shared__ unsigned short sAH[8][16 * 128];    // per-wave A tile, reused for hidden
  __shared__ int            sSrc[8][16];
  __shared__ int            sDst[8][16];

  const unsigned int* gW0 = (const unsigned int*)(ws + WS_W0F);
  const unsigned int* gW1 = (const unsigned int*)(ws + WS_W1F);
  for (int i = threadIdx.x; i < 128 * 64; i += 256) sW0[i] = gW0[i];
  for (int i = threadIdx.x; i < 96 * 64;  i += 256) sW1[i] = gW1[i];
  if (threadIdx.x < 128) sB0[threadIdx.x] = fb0[threadIdx.x];
  if (threadIdx.x < 96)  sB1[threadIdx.x] = fb1[threadIdx.x];
  __syncthreads();   // weights staged once per (persistent) block

  const int lane = threadIdx.x & 31;
  const int wave = threadIdx.x >> 5;
  const int half = lane >> 4;

  const float*        delta = (const float*)(ws + WS_DELTA);
  float*              agg   = (float*)(ws + WS_AGG);
  const unsigned int* xbf   = (const unsigned int*)(ws + WS_XBF);  // 48 u32 / node
  const unsigned int* aRow  = (const unsigned int*)&sAH[wave][(lane & 15) * 128];

  for (int t = blockIdx.x; t < EDGE_TILES; t += EDGE_GRID) {
    const long long eBase = (long long)t * 128 + wave * 16;

    // prefetch next iteration's edge indices (global_prefetch_b8)
    const long long pBase = (long long)(t + EDGE_GRID) * 128 + wave * 16;
    if (pBase < N_EDGES && lane == 0) {
      __builtin_prefetch(&ei[pBase], 0, 1);
      __builtin_prefetch(&ei[(long long)N_EDGES + pBase], 0, 1);
    }

    if (lane < 16) {
      sSrc[wave][lane] = (int)ei[eBase + lane];
      sDst[wave][lane] = (int)ei[(long long)N_EDGES + eBase + lane];
    }
    // features [0..95] = x_bf16[src]
    for (int i = lane; i < 16 * 48; i += 32) {
      int m = i / 48, j = i - m * 48;
      int s = sSrc[wave][m];
      ((unsigned int*)&sAH[wave][m * 128])[j] = xbf[(long long)s * 48 + j];
    }
    // features [96..98] = rel, [99..127] = 0
    if (lane < 16) {
      int m = lane;
      int s = sSrc[wave][m], d = sDst[wave][m];
      float r0 = pos[s * 3 + 0] - pos[d * 3 + 0] + delta[d * 3 + 0];
      float r1 = pos[s * 3 + 1] - pos[d * 3 + 1] + delta[d * 3 + 1];
      float r2 = pos[s * 3 + 2] - pos[d * 3 + 2] + delta[d * 3 + 2];
      unsigned int* rowu = (unsigned int*)&sAH[wave][m * 128];
      rowu[48] = (unsigned int)bf16b(r0) | ((unsigned int)bf16b(r1) << 16);
      rowu[49] = (unsigned int)bf16b(r2);
#pragma unroll
      for (int j = 50; j < 64; ++j) rowu[j] = 0u;
    }

    // layer 1: K=128 (padded), N=128
    v8f acc[8];
#pragma unroll
    for (int n = 0; n < 8; ++n)
#pragma unroll
      for (int r = 0; r < 8; ++r) acc[n][r] = 0.0f;
#pragma unroll
    for (int kc = 0; kc < 4; ++kc) {
      v16bf a = ldfrag(aRow, kc * 32, half);
#pragma unroll
      for (int n = 0; n < 8; ++n) {
        v16bf b = ldfrag(&sW0[(n * 16 + (lane & 15)) * 64], kc * 32, half);
        acc[n] = __builtin_amdgcn_wmma_f32_16x16x32_bf16(false, a, false, b,
                                                         (short)0, acc[n], false, false);
      }
    }
    // bias + relu -> sAH (A of layer 2); same-wave DS ops are in-order
#pragma unroll
    for (int n = 0; n < 8; ++n) {
      int f = n * 16 + (lane & 15);
      float bb = sB0[f];
#pragma unroll
      for (int r = 0; r < 8; ++r) {
        float h = fmaxf(acc[n][r] + bb, 0.0f);
        sAH[wave][(r + 8 * half) * 128 + f] = bf16b(h);
      }
    }
    // layer 2: K=128, N=96
    v8f acc2[6];
#pragma unroll
    for (int n = 0; n < 6; ++n)
#pragma unroll
      for (int r = 0; r < 8; ++r) acc2[n][r] = 0.0f;
#pragma unroll
    for (int kc = 0; kc < 4; ++kc) {
      v16bf a = ldfrag(aRow, kc * 32, half);
#pragma unroll
      for (int n = 0; n < 6; ++n) {
        v16bf b = ldfrag(&sW1[(n * 16 + (lane & 15)) * 64], kc * 32, half);
        acc2[n] = __builtin_amdgcn_wmma_f32_16x16x32_bf16(false, a, false, b,
                                                          (short)0, acc2[n], false, false);
      }
    }
    // scatter-max into agg
#pragma unroll
    for (int n = 0; n < 6; ++n) {
      int f = n * 16 + (lane & 15);
      float bb = sB1[f];
#pragma unroll
      for (int r = 0; r < 8; ++r) {
        int m = r + 8 * half;
        atomicMaxF(&agg[(long long)sDst[wave][m] * FEAT + f], acc2[n][r] + bb);
      }
    }
  }
}

// ---------------- node update: out = x + mlp_g(agg), WMMA ----------------
__global__ void __launch_bounds__(256) node_kernel(
    const float* __restrict__ x,
    const float* __restrict__ gb0, const float* __restrict__ gb1,
    float* __restrict__ out, char* __restrict__ ws) {
  __shared__ unsigned int   sW0[128 * 64];
  __shared__ unsigned int   sW1[96 * 64];
  __shared__ float          sB0[128];
  __shared__ float          sB1[96];
  __shared__ unsigned short sAH[8][16 * 128];

  const unsigned int* gW0 = (const unsigned int*)(ws + WS_W0G);
  const unsigned int* gW1 = (const unsigned int*)(ws + WS_W1G);
  for (int i = threadIdx.x; i < 128 * 64; i += 256) sW0[i] = gW0[i];
  for (int i = threadIdx.x; i < 96 * 64;  i += 256) sW1[i] = gW1[i];
  if (threadIdx.x < 128) sB0[threadIdx.x] = gb0[threadIdx.x];
  if (threadIdx.x < 96)  sB1[threadIdx.x] = gb1[threadIdx.x];
  __syncthreads();

  const int lane = threadIdx.x & 31;
  const int wave = threadIdx.x >> 5;
  const int half = lane >> 4;
  const int tile = blockIdx.x * 8 + wave;
  if (tile * 16 >= N_NODES) return;   // after the only barrier
  const long long nBase = (long long)tile * 16;

  const float* agg = (const float*)(ws + WS_AGG);

  // stage A: sanitized agg -> bf16 (only k<96 is ever read in layer 1)
  for (int i = lane; i < 16 * 48; i += 32) {
    int m = i / 48, j = i - m * 48;
    const float* ar = agg + (nBase + m) * FEAT + j * 2;
    float v0 = ar[0], v1 = ar[1];
    if (__float_as_uint(v0) == 0xFF800000u) v0 = 0.0f;
    if (__float_as_uint(v1) == 0xFF800000u) v1 = 0.0f;
    ((unsigned int*)&sAH[wave][m * 128])[j] =
        (unsigned int)bf16b(v0) | ((unsigned int)bf16b(v1) << 16);
  }

  const unsigned int* aRow = (const unsigned int*)&sAH[wave][(lane & 15) * 128];

  v8f acc[8];
#pragma unroll
  for (int n = 0; n < 8; ++n)
#pragma unroll
    for (int r = 0; r < 8; ++r) acc[n][r] = 0.0f;
#pragma unroll
  for (int kc = 0; kc < 3; ++kc) {  // K=96, skip zero chunk
    v16bf a = ldfrag(aRow, kc * 32, half);
#pragma unroll
    for (int n = 0; n < 8; ++n) {
      v16bf b = ldfrag(&sW0[(n * 16 + (lane & 15)) * 64], kc * 32, half);
      acc[n] = __builtin_amdgcn_wmma_f32_16x16x32_bf16(false, a, false, b,
                                                       (short)0, acc[n], false, false);
    }
  }
#pragma unroll
  for (int n = 0; n < 8; ++n) {
    int f = n * 16 + (lane & 15);
    float bb = sB0[f];
#pragma unroll
    for (int r = 0; r < 8; ++r) {
      float h = fmaxf(acc[n][r] + bb, 0.0f);
      sAH[wave][(r + 8 * half) * 128 + f] = bf16b(h);
    }
  }
  v8f acc2[6];
#pragma unroll
  for (int n = 0; n < 6; ++n)
#pragma unroll
    for (int r = 0; r < 8; ++r) acc2[n][r] = 0.0f;
#pragma unroll
  for (int kc = 0; kc < 4; ++kc) {
    v16bf a = ldfrag(aRow, kc * 32, half);
#pragma unroll
    for (int n = 0; n < 6; ++n) {
      v16bf b = ldfrag(&sW1[(n * 16 + (lane & 15)) * 64], kc * 32, half);
      acc2[n] = __builtin_amdgcn_wmma_f32_16x16x32_bf16(false, a, false, b,
                                                        (short)0, acc2[n], false, false);
    }
  }
#pragma unroll
  for (int n = 0; n < 6; ++n) {
    int f = n * 16 + (lane & 15);
    float bb = sB1[f];
#pragma unroll
    for (int r = 0; r < 8; ++r) {
      long long node = nBase + r + 8 * half;
      out[node * FEAT + f] = x[node * FEAT + f] + acc2[n][r] + bb;
    }
  }
}

extern "C" void kernel_launch(void* const* d_in, const int* in_sizes, int n_in,
                              void* d_out, int out_size, void* d_ws, size_t ws_size,
                              hipStream_t stream) {
  (void)in_sizes; (void)n_in; (void)out_size; (void)ws_size;
  const float*     x   = (const float*)d_in[0];
  const float*     pos = (const float*)d_in[1];
  const long long* ei  = (const long long*)d_in[2];
  const float* hw0 = (const float*)d_in[3];
  const float* hb0 = (const float*)d_in[4];
  const float* hw1 = (const float*)d_in[5];
  const float* hb1 = (const float*)d_in[6];
  const float* fw0 = (const float*)d_in[7];
  const float* fb0 = (const float*)d_in[8];
  const float* fw1 = (const float*)d_in[9];
  const float* fb1 = (const float*)d_in[10];
  const float* gw0 = (const float*)d_in[11];
  const float* gb0 = (const float*)d_in[12];
  const float* gw1 = (const float*)d_in[13];
  const float* gb1 = (const float*)d_in[14];
  float* out = (float*)d_out;
  char*  ws  = (char*)d_ws;

  long long prepTotal = 2LL * N_NODES * FEAT + 2 * (128 * 128) + 2 * (96 * 128);
  int prepBlocks = (int)((prepTotal + 255) / 256);
  prep_kernel<<<prepBlocks, 256, 0, stream>>>(x, fw0, fw1, gw0, gw1, ws);
  delta_kernel<<<(N_NODES + 255) / 256, 256, 0, stream>>>(x, hw0, hb0, hw1, hb1, ws);
  edge_kernel<<<EDGE_GRID, 256, 0, stream>>>(pos, ei, fb0, fb1, ws);
  node_kernel<<<(N_NODES / 16 + 7) / 8, 256, 0, stream>>>(x, gb0, gb1, out, ws);
}
